// INNfoil_3959959847184
// MI455X (gfx1250) — matrix operand order repaired and checked by reading, not compiled, not run on hardware
//
#include <hip/hip_runtime.h>
#include <hip/hip_bf16.h>
#include <cstdint>
#include <cstddef>

typedef __bf16 bf16_t;
typedef bf16_t v16bf __attribute__((ext_vector_type(16)));
typedef bf16_t v8bf  __attribute__((ext_vector_type(8)));
typedef bf16_t v4bf  __attribute__((ext_vector_type(4)));
typedef float  v8f   __attribute__((ext_vector_type(8)));
typedef unsigned int u32x4 __attribute__((ext_vector_type(4)));
typedef int i32x4 __attribute__((ext_vector_type(4)));
typedef int i32x8 __attribute__((ext_vector_type(8)));

static constexpr int kN   = 65536;
static constexpr int kC   = 256;
static constexpr int kC2  = 128;
static constexpr int kH   = 384;
static constexpr int kL   = 15;
static constexpr int kMT  = 64;    // rows per block
static constexpr int kTPB = 256;   // 8 wave32 waves

// dynamic LDS layout (bytes); WGP has 320KB, we use 288KB:
// double-buffered weights so TDM DMA overlaps WMMA compute.
static constexpr int XA_OFF = 0;                         // 64x128 bf16  = 16 KB
static constexpr int HS_OFF = XA_OFF + kMT * kC2 * 2;    // 64x384 bf16  = 48 KB
static constexpr int WA_OFF = HS_OFF + kMT * kH  * 2;    // weight buf A = 96 KB
static constexpr int WB_OFF = WA_OFF + kC2 * kH  * 2;    // weight buf B = 96 KB
static constexpr int SB_OFF = WB_OFF + kC2 * kH  * 2;    // 64x128 f32   = 32 KB
static constexpr int SMEM_BYTES = SB_OFF + kMT * kC2 * 4; // 288 KB

#if defined(__has_builtin)
#if __has_builtin(__builtin_amdgcn_tensor_load_to_lds) && \
    __has_builtin(__builtin_amdgcn_s_wait_tensorcnt)
#define ATHENA_USE_TDM 1
#endif
#endif

// ------------------------------------------------------------- weight cvt
// Converts f32 weights to bf16 AND transposes each layer's [K][N] matrix to
// [N][K] so WMMA A-fragments (weights) are K-contiguous per feature row.
__global__ void cvt_w_bf16_T(const float* __restrict__ src,
                             bf16_t* __restrict__ dst,
                             int K, int Nn, int total) {
  int i = blockIdx.x * blockDim.x + threadIdx.x;
  if (i >= total) return;
  const int per = K * Nn;
  const int l = i / per;
  const int r = i - l * per;
  const int k = r / Nn;
  const int n = r - k * Nn;
  dst[(size_t)l * per + (size_t)n * K + k] = (bf16_t)src[i];
}

// ------------------------------------------------------------- LDS staging
// Issue an async weight stage into LDS at lds_off. With TDM: wave 0 issues a
// single-descriptor 1-row tile (49152 x 2B) on the Tensor Data Mover and
// returns immediately (completion via TENSORcnt). Fallback: synchronous copy.
__device__ __forceinline__ void tdm_issue(char* __restrict__ smem,
                                          unsigned lds_off,
                                          const bf16_t* __restrict__ src,
                                          int nelem, int tid) {
#if defined(ATHENA_USE_TDM)
  if (tid < 32) {
    const unsigned long long ga = (unsigned long long)(uintptr_t)src;
    u32x4 g0;
    g0[0] = 1u;                                   // count=1, user descriptor
    g0[1] = lds_off;                              // lds_addr (bytes)
    g0[2] = (unsigned)(ga & 0xffffffffu);         // global_addr[31:0]
    g0[3] = (unsigned)((ga >> 32) & 0x01ffffffu)  // global_addr[56:32]
            | (2u << 30);                         // type=2 ("image")
    const unsigned ne = (unsigned)nelem;          // 49152 fits 16-bit tile dim
    i32x8 g1;
    g1[0] = (int)(1u << 16);                      // data_size=2B, mask=0
    g1[1] = (int)((ne & 0xffffu) << 16);          // tensor_dim0[15:0]
    g1[2] = (int)((ne >> 16) | (1u << 16));       // tensor_dim0[31:16], dim1=1
    g1[3] = (int)((ne & 0xffffu) << 16);          // tile_dim0
    g1[4] = 1;                                    // tile_dim1 = 1
    g1[5] = (int)ne;                              // tensor_dim0_stride[31:0]
    g1[6] = 0;
    g1[7] = 0;
    i32x4 z4 = {0, 0, 0, 0};
#if defined(__clang_major__) && (__clang_major__ >= 23)
    i32x8 z8 = {0, 0, 0, 0, 0, 0, 0, 0};
    __builtin_amdgcn_tensor_load_to_lds(g0, g1, z4, z4, z8, 0);
#else
    __builtin_amdgcn_tensor_load_to_lds(g0, g1, z4, z4, 0);
#endif
  }
#else
  // synchronous fallback copy (no overlap)
  const uint4* s = (const uint4*)src;
  uint4* d = (uint4*)(smem + lds_off);
  const int nv = nelem >> 3;
  for (int i = tid; i < nv; i += kTPB) d[i] = s[i];
#endif
}

template <int PENDING>
__device__ __forceinline__ void tdm_wait(int tid) {
#if defined(ATHENA_USE_TDM)
  if (tid < 32) __builtin_amdgcn_s_wait_tensorcnt(PENDING);
#endif
}

// ------------------------------------------------------------ frag loaders
// A-fragment (16x32 bf16 weights, row-major [feature][k]): lane holds
// K {0..15} (+16 for upper half-wave) -> two 16B LDS vector loads.
__device__ __forceinline__ v16bf load_a_frag(const bf16_t* __restrict__ p) {
  v8bf lo = *(const v8bf*)(p);
  v8bf hi = *(const v8bf*)(p + 8);
  return __builtin_shufflevector(lo, hi, 0, 1, 2, 3, 4, 5, 6, 7,
                                 8, 9, 10, 11, 12, 13, 14, 15);
}

__device__ __forceinline__ float leaky(float v) { return fmaxf(v, 0.2f * v); }

// ------------------------------------- GEMM1': h^T = leaky(W0^T @ x^T + b0)
// A = W0^T (384x128 in LDS [feature][k]); B = x^T from Xa[sample][k];
// D tiles: M=feature, N=sample.  24 f-tiles x 4 m-tiles = 96 / 8 waves.
// Software-pipelined: next f-tile's 4 A-frags load under current WMMA chain.
__device__ __forceinline__ void mlp_front(const bf16_t* __restrict__ Xa,
                                          const bf16_t* __restrict__ Wl,
                                          const float*  __restrict__ b0,
                                          bf16_t* __restrict__ Hs,
                                          int wid, int lane) {
  const int mt  = wid & 3;                 // sample tile
  const int fh  = wid >> 2;                // feature half: f-tiles fh*12..+12
  const int m   = mt * 16 + (lane & 15);   // this lane's sample column
  const int hi  = (lane >> 4) & 1;
  const int klo = hi * 16;
  const int frow = (lane & 15);            // A-row within f-tile

  v16bf bx[4];                             // activation B-frags, reused 12x
#pragma unroll
  for (int kc = 0; kc < 4; ++kc)
    bx[kc] = *(const v16bf*)(Xa + m * kC2 + kc * 32 + hi * 16);

  v16bf aa[4];                             // current f-tile A-frags
#pragma unroll
  for (int kc = 0; kc < 4; ++kc)
    aa[kc] = load_a_frag(Wl + ((fh * 12) * 16 + frow) * kC2 + kc * 32 + klo);

  for (int j = 0; j < 12; ++j) {
    const int ft = fh * 12 + j;
    const int fbase = ft * 16 + hi * 8;    // 8 consecutive features per lane
    const float4 bva = *(const float4*)(b0 + fbase);
    const float4 bvb = *(const float4*)(b0 + fbase + 4);
    v8f acc = {bva.x, bva.y, bva.z, bva.w, bvb.x, bvb.y, bvb.z, bvb.w};

    // prefetch next tile's A-frags (last iteration re-fetches itself)
    const int ftn = (j + 1 < 12) ? ft + 1 : ft;
    v16bf an[4];
#pragma unroll
    for (int kc = 0; kc < 4; ++kc)
      an[kc] = load_a_frag(Wl + (ftn * 16 + frow) * kC2 + kc * 32 + klo);

#pragma unroll
    for (int kc = 0; kc < 4; ++kc)
      acc = __builtin_amdgcn_wmma_f32_16x16x32_bf16(false, aa[kc], false, bx[kc],
                                                    (short)0, acc, false, false);
    v8bf hv;
#pragma unroll
    for (int r = 0; r < 8; ++r) hv[r] = (bf16_t)leaky(acc[r]);
    *(v8bf*)(Hs + m * kH + fbase) = hv;    // one 16B store per tile

#pragma unroll
    for (int kc = 0; kc < 4; ++kc) aa[kc] = an[kc];
  }
}

// ------------------------------------- GEMM2': y^T = leaky(W1^T @ h^T + b1)
// A = W1^T (128x384 in LDS [feature][k]); B = h^T from Hs[sample][k];
// D tiles: M=out-feature (8 tiles), N=sample (4 tiles) -> 4 per wave.
// Software-pipelined over K: next kc's B-frag + 4 A-frags load under WMMAs.
__device__ __forceinline__ void mlp_back(const bf16_t* __restrict__ Hs,
                                         const bf16_t* __restrict__ Wl,
                                         const float*  __restrict__ b1,
                                         int wid, int lane, v8f acc[4]) {
  const int mt  = wid & 3;
  const int fh  = wid >> 2;                // out-feature tiles fh*4..+4
  const int m   = mt * 16 + (lane & 15);
  const int hi  = (lane >> 4) & 1;
  const int klo = hi * 16;
  const int frow = (lane & 15);

#pragma unroll
  for (int j = 0; j < 4; ++j) {
    const int fbase = (fh * 4 + j) * 16 + hi * 8;
    const float4 bva = *(const float4*)(b1 + fbase);
    const float4 bvb = *(const float4*)(b1 + fbase + 4);
    v8f t = {bva.x, bva.y, bva.z, bva.w, bvb.x, bvb.y, bvb.z, bvb.w};
    acc[j] = t;
  }

  v16bf bcur = *(const v16bf*)(Hs + m * kH + hi * 16);
  v16bf av[4];
#pragma unroll
  for (int j = 0; j < 4; ++j)
    av[j] = load_a_frag(Wl + ((fh * 4 + j) * 16 + frow) * kH + klo);

  for (int kc = 0; kc < 12; ++kc) {
    const int kn = (kc + 1 < 12) ? kc + 1 : kc;    // prefetch next K chunk
    v16bf bn = *(const v16bf*)(Hs + m * kH + kn * 32 + hi * 16);
    v16bf an[4];
#pragma unroll
    for (int j = 0; j < 4; ++j)
      an[j] = load_a_frag(Wl + ((fh * 4 + j) * 16 + frow) * kH + kn * 32 + klo);

#pragma unroll
    for (int j = 0; j < 4; ++j)
      acc[j] = __builtin_amdgcn_wmma_f32_16x16x32_bf16(false, av[j], false, bcur,
                                                       (short)0, acc[j], false, false);
    bcur = bn;
#pragma unroll
    for (int j = 0; j < 4; ++j) av[j] = an[j];
  }
#pragma unroll
  for (int j = 0; j < 4; ++j)
#pragma unroll
    for (int r = 0; r < 8; ++r) acc[j][r] = leaky(acc[j][r]);
}

// --------------------------------------------------------- half-layer kernel
// out = out * exp(mlp_s(in)) + mlp_t(in)   (in/out are 128-wide halves of x)
__global__ void __launch_bounds__(kTPB)
coupling_half(float* __restrict__ x,
              const bf16_t* __restrict__ W0s, const float* __restrict__ b0s,
              const bf16_t* __restrict__ W1s, const float* __restrict__ b1s,
              const bf16_t* __restrict__ W0t, const float* __restrict__ b0t,
              const bf16_t* __restrict__ W1t, const float* __restrict__ b1t,
              int in_off, int out_off) {
  extern __shared__ char smem[];
  bf16_t* Xa = (bf16_t*)(smem + XA_OFF);
  bf16_t* Hs = (bf16_t*)(smem + HS_OFF);
  bf16_t* Wa = (bf16_t*)(smem + WA_OFF);
  bf16_t* Wb = (bf16_t*)(smem + WB_OFF);
  float*  Sb = (float*)(smem + SB_OFF);

  const int tid  = threadIdx.x;
  const int lane = tid & 31;
  const int wid  = tid >> 5;
  const int row0 = blockIdx.x * kMT;

  __builtin_prefetch((const void*)W0t, 0, 1);
  __builtin_prefetch((const void*)W1t, 0, 1);

  // kick off both scale-MLP weight DMAs up front (in-order per wave)
  tdm_issue(smem, WA_OFF, W0s, kC2 * kH, tid);
  tdm_issue(smem, WB_OFF, W1s, kH * kC2, tid);

  // stage input half-tile -> LDS bf16, row-major [sample][k]
  // (overlaps with the weight DMA)
  for (int i = tid; i < kMT * kC2 / 4; i += kTPB) {
    const int r = i >> 5;
    const int c = (i & 31) << 2;
    const float4 v = *(const float4*)&x[(size_t)(row0 + r) * kC + in_off + c];
    v4bf p;
    p[0] = (bf16_t)v.x; p[1] = (bf16_t)v.y;
    p[2] = (bf16_t)v.z; p[3] = (bf16_t)v.w;
    *(v4bf*)(Xa + r * kC2 + c) = p;          // 8B packed store
  }
  tdm_wait<1>(tid);                          // W0s landed (W1s may be in flight)
  __syncthreads();

  const int fh = wid >> 2;
  const int m  = (wid & 3) * 16 + (lane & 15);
  const int hi = (lane >> 4) & 1;

  // -------- scale MLP --------
  mlp_front(Xa, Wa, b0s, Hs, wid, lane);     // compute overlaps W1s DMA
  __syncthreads();                           // all readers of Wa done
  tdm_issue(smem, WA_OFF, W0t, kC2 * kH, tid); // W0t streams in behind W1s
  tdm_wait<1>(tid);                          // W1s landed
  __syncthreads();
  {
    v8f acc[4];
    mlp_back(Hs, Wb, b1s, wid, lane, acc);   // compute overlaps W0t DMA
#pragma unroll
    for (int j = 0; j < 4; ++j) {
      const int fbase = (fh * 4 + j) * 16 + hi * 8;
      float4 lo = {acc[j][0], acc[j][1], acc[j][2], acc[j][3]};
      float4 hi4 = {acc[j][4], acc[j][5], acc[j][6], acc[j][7]};
      *(float4*)(Sb + m * kC2 + fbase)     = lo;   // 16B LDS stores
      *(float4*)(Sb + m * kC2 + fbase + 4) = hi4;
    }
  }
  __syncthreads();                           // readers of Wb + Hs done
  tdm_issue(smem, WB_OFF, W1t, kH * kC2, tid); // W1t streams in behind W0t
  tdm_wait<1>(tid);                          // W0t landed
  __syncthreads();

  // -------- translate MLP --------
  mlp_front(Xa, Wa, b0t, Hs, wid, lane);     // compute overlaps W1t DMA
  tdm_wait<0>(tid);                          // W1t landed
  __syncthreads();
  {
    v8f acc[4];
    mlp_back(Hs, Wb, b1t, wid, lane, acc);
    float* gp0 = &x[(size_t)(row0 + m) * kC + out_off];
#pragma unroll
    for (int j = 0; j < 4; ++j) {
      const int fbase = (fh * 4 + j) * 16 + hi * 8;
      const float4 s0 = *(const float4*)(Sb + m * kC2 + fbase);
      const float4 s1 = *(const float4*)(Sb + m * kC2 + fbase + 4);
      float4 x0 = *(const float4*)(gp0 + fbase);
      float4 x1 = *(const float4*)(gp0 + fbase + 4);
      x0.x = x0.x * __expf(s0.x) + acc[j][0];
      x0.y = x0.y * __expf(s0.y) + acc[j][1];
      x0.z = x0.z * __expf(s0.z) + acc[j][2];
      x0.w = x0.w * __expf(s0.w) + acc[j][3];
      x1.x = x1.x * __expf(s1.x) + acc[j][4];
      x1.y = x1.y * __expf(s1.y) + acc[j][5];
      x1.z = x1.z * __expf(s1.z) + acc[j][6];
      x1.w = x1.w * __expf(s1.w) + acc[j][7];
      *(float4*)(gp0 + fbase)     = x0;            // 16B global stores
      *(float4*)(gp0 + fbase + 4) = x1;
    }
  }
}

// ------------------------------------------------------------------- launch
extern "C" void kernel_launch(void* const* d_in, const int* in_sizes, int n_in,
                              void* d_out, int out_size, void* d_ws,
                              size_t ws_size, hipStream_t stream) {
  (void)in_sizes; (void)out_size; (void)ws_size;
  if (n_in < 17) return;

  const float* x_in = (const float*)d_in[0];
  float* x = (float*)d_out;

  // d_in: 0:x, then per nm in [s0,t0,s1,t1]: W0(1,5,9,13) b0(2,6,10,14)
  //       W1(3,7,11,15) b1(4,8,12,16)
  bf16_t* ws = (bf16_t*)d_ws;
  const size_t PER = (size_t)kL * kC2 * kH;  // 737280 elems per weight tensor
  // ws layout (each transposed to [n][k] per layer):
  // [W0_s0, W1_s0, W0_t0, W1_t0, W0_s1, W1_s1, W0_t1, W1_t1]
  const int src_idx[8] = {1, 3, 5, 7, 9, 11, 13, 15};
  const int cvt_blocks = (int)((PER + kTPB - 1) / kTPB);
  for (int a = 0; a < 8; ++a) {
    const bool isW0 = (a & 1) == 0;   // even slots are W0 (K=128,N=384)
    cvt_w_bf16_T<<<cvt_blocks, kTPB, 0, stream>>>(
        (const float*)d_in[src_idx[a]], ws + (size_t)a * PER,
        isW0 ? kC2 : kH, isW0 ? kH : kC2, (int)PER);
  }

  hipMemcpyAsync(x, x_in, (size_t)kN * kC * sizeof(float),
                 hipMemcpyDeviceToDevice, stream);

  const dim3 grid(kN / kMT);
  for (int l = 0; l < kL; ++l) {
    const size_t wOff = (size_t)l * kC2 * kH;
    // x0 = x0 * exp(mlp_s1(x1)) + mlp_t1(x1)
    coupling_half<<<grid, kTPB, SMEM_BYTES, stream>>>(
        x,
        ws + 4 * PER + wOff, (const float*)d_in[10] + (size_t)l * kH,
        ws + 5 * PER + wOff, (const float*)d_in[12] + (size_t)l * kC2,
        ws + 6 * PER + wOff, (const float*)d_in[14] + (size_t)l * kH,
        ws + 7 * PER + wOff, (const float*)d_in[16] + (size_t)l * kC2,
        kC2, 0);
    // x1 = x1 * exp(mlp_s0(x0)) + mlp_t0(x0)
    coupling_half<<<grid, kTPB, SMEM_BYTES, stream>>>(
        x,
        ws + 0 * PER + wOff, (const float*)d_in[2] + (size_t)l * kH,
        ws + 1 * PER + wOff, (const float*)d_in[4] + (size_t)l * kC2,
        ws + 2 * PER + wOff, (const float*)d_in[6] + (size_t)l * kH,
        ws + 3 * PER + wOff, (const float*)d_in[8] + (size_t)l * kC2,
        0, kC2);
  }
}
